// Attention_seq2seq_18339510354538
// MI455X (gfx1250) — compile-verified, matrix operand order checked by reading, not run on hardware
//
#include <hip/hip_runtime.h>
#include <hip/hip_bf16.h>

// ---------------------------------------------------------------------------
// Seq2seq attention LSTM for MI455X (gfx1250, wave32, WMMA 16x16x32 bf16).
//
// All GEMMs run through v_wmma_f32_16x16x32_bf16 (f32 accum). Weights are
// L2-resident (~32MB bf16 vs 192MB L2), so the sequential LSTM chain is
// latency-bound; everything time-parallel (input-gate precompute, enc_proj,
// the 8128x16000x512 vocab projection) is hoisted into large parallel GEMMs.
// ---------------------------------------------------------------------------

typedef __bf16 bf16_t;
typedef __attribute__((ext_vector_type(16))) __bf16 v16bf;
typedef __attribute__((ext_vector_type(8)))  __bf16 v8bf;
typedef __attribute__((ext_vector_type(8)))  float  v8f;

#define BB 64      // batch
#define SS 256     // src len
#define TT 128     // tgt len
#define TD 127     // tgt steps (T-1)
#define EE 256     // embed
#define HH 512     // hidden
#define GG 2048    // 4H gates
#define DH 1024    // 2H
#define VT 16000   // tgt vocab

union FragU { v16bf v; v8bf h[2]; };

// Load a 16x32 bf16 fragment from row-major [rows, K] memory (row stride ld).
// Layout per ISA 7.12.2 (16-bit A / B-as-[N,K]):
//   lane<16 : row=lane,    K = {k0..k0+7, k0+16..k0+23}
//   lane>=16: row=lane-16, K = {k0+8..k0+15, k0+24..k0+31}
__device__ __forceinline__ v16bf load_frag(const bf16_t* base, int ld, int k0) {
  const int lane = threadIdx.x & 31;
  const int r    = lane & 15;
  const int hi   = (lane >> 4) << 3;  // 0 or 8
  const bf16_t* p = base + (size_t)r * ld + k0 + hi;
  FragU f;
  f.h[0] = *(const v8bf*)(p);
  f.h[1] = *(const v8bf*)(p + 16);
  return f.v;
}

__device__ __forceinline__ float sigf(float x) { return 1.0f / (1.0f + expf(-x)); }

// ---------------------------------------------------------------------------
// Generic GEMM: Y[M,N] = A[M,K] @ W[N,K]^T, bf16 inputs, f32 accumulate.
// Block = 256 threads (8 waves). Each block: one 16-row M tile, 8 N tiles
// (one per wave). A chunk (16 x 256) staged in LDS (+8 pad, bank-safe).
// Modes: 0=bf16 out, 1=f32 out, 2=f32+bf16 out, 3=tanh->bf16, 4=f32 with
// (t,b)->out[b][t][v] remap for the final vocab projection.
// ---------------------------------------------------------------------------
#define KC 256
__global__ void __launch_bounds__(256)
gemm_bf16(const bf16_t* __restrict__ A, int lda,
          const bf16_t* __restrict__ Bw, int ldb,
          int K, int mode,
          float* __restrict__ outF, int ldo,
          bf16_t* __restrict__ outB, int ldob)
{
  __shared__ bf16_t As[16][KC + 8];
  const int m0   = blockIdx.y * 16;
  const int wave = threadIdx.x >> 5;
  const int n0   = blockIdx.x * 128 + wave * 16;
  const int arow = threadIdx.x >> 4;
  const int acol = (threadIdx.x & 15) * 16;
  v8f acc = {};

  for (int kc = 0; kc < K; kc += KC) {
    const bf16_t* ap = A + (size_t)(m0 + arow) * lda + kc + acol;
    *(v8bf*)&As[arow][acol]     = *(const v8bf*)(ap);
    *(v8bf*)&As[arow][acol + 8] = *(const v8bf*)(ap + 8);
    __syncthreads();
    const bf16_t* bp = Bw + (size_t)n0 * ldb + kc;
    for (int kk = 0; kk < KC; kk += 32) {
      v16bf a = load_frag(&As[0][0], KC + 8, kk);
      v16bf b = load_frag(bp, ldb, kk);
      acc = __builtin_amdgcn_wmma_f32_16x16x32_bf16(false, a, false, b,
                                                    (short)0, acc, false, false);
    }
    __syncthreads();
  }

  const int lane = threadIdx.x & 31;
  const int n    = lane & 15;
  const int mo   = (lane >> 4) * 8;
  if (mode == 0) {
    for (int v = 0; v < 8; ++v)
      outB[(size_t)(m0 + mo + v) * ldob + n0 + n] = (bf16_t)acc[v];
  } else if (mode == 1) {
    for (int v = 0; v < 8; ++v)
      outF[(size_t)(m0 + mo + v) * ldo + n0 + n] = acc[v];
  } else if (mode == 2) {
    for (int v = 0; v < 8; ++v) {
      outF[(size_t)(m0 + mo + v) * ldo  + n0 + n] = acc[v];
      outB[(size_t)(m0 + mo + v) * ldob + n0 + n] = (bf16_t)acc[v];
    }
  } else if (mode == 3) {
    for (int v = 0; v < 8; ++v)
      outB[(size_t)(m0 + mo + v) * ldob + n0 + n] = (bf16_t)tanhf(acc[v]);
  } else { // mode 4: rows are m = t*BB + b ; out[b][t][v]
    for (int v = 0; v < 8; ++v) {
      const int m = m0 + mo + v;
      const int b = m & 63;
      const int t = m >> 6;
      outF[(size_t)b * TD * VT + (size_t)t * VT + n0 + n] = acc[v];
    }
  }
}

// ---------------------------------------------------------------------------
// Fused per-step LSTM: g = A1@W1^T (+ A2@W2^T) + Gx + bih + bhh, then cell.
// Block = 512 threads (16 waves), grid = 4 (16 batch rows per block).
// A1 = h_prev (bf16 rows), A2 = o_prev (decoder). null => zero contribution.
// Each block owns its 16 batch rows end-to-end (no cross-block hazards).
// ---------------------------------------------------------------------------
__global__ void __launch_bounds__(512)
lstm_step(const bf16_t* __restrict__ Gx,
          const bf16_t* __restrict__ W1, int ldb1,
          const bf16_t* __restrict__ A1, int lda1,
          const bf16_t* __restrict__ W2, int ldb2,
          const bf16_t* __restrict__ A2, int lda2,
          const float* __restrict__ bih, const float* __restrict__ bhh,
          float* __restrict__ cst, float* __restrict__ hf32,
          bf16_t* __restrict__ hbf, int ldhb,
          float* __restrict__ gbuf)
{
  __shared__ bf16_t A1s[16][HH + 8];
  __shared__ bf16_t A2s[16][HH + 8];
  const int m0  = blockIdx.x * 16;
  const int row = threadIdx.x >> 5;
  const int col = (threadIdx.x & 31) * 16;
  const bool useA1 = (A1 != nullptr);
  const bool useA2 = (W2 != nullptr) && (A2 != nullptr);

  if (useA1) {
    const bf16_t* p = A1 + (size_t)(m0 + row) * lda1 + col;
    *(v8bf*)&A1s[row][col]     = *(const v8bf*)(p);
    *(v8bf*)&A1s[row][col + 8] = *(const v8bf*)(p + 8);
  }
  if (useA2) {
    const bf16_t* p = A2 + (size_t)(m0 + row) * lda2 + col;
    *(v8bf*)&A2s[row][col]     = *(const v8bf*)(p);
    *(v8bf*)&A2s[row][col + 8] = *(const v8bf*)(p + 8);
  }
  __syncthreads();

  const int wave = threadIdx.x >> 5;
  const int lane = threadIdx.x & 31;
  const int nloc = lane & 15;
  const int mo   = (lane >> 4) * 8;

  for (int i = 0; i < 8; ++i) {          // 128 N-tiles / 16 waves
    const int n0 = (wave + i * 16) * 16;
    v8f acc = {};
    if (useA1) {
      const bf16_t* bp = W1 + (size_t)n0 * ldb1;
      for (int k = 0; k < HH; k += 32) {
        v16bf a = load_frag(&A1s[0][0], HH + 8, k);
        v16bf b = load_frag(bp, ldb1, k);
        acc = __builtin_amdgcn_wmma_f32_16x16x32_bf16(false, a, false, b,
                                                      (short)0, acc, false, false);
      }
    }
    if (useA2) {
      const bf16_t* bp = W2 + (size_t)n0 * ldb2;
      for (int k = 0; k < HH; k += 32) {
        v16bf a = load_frag(&A2s[0][0], HH + 8, k);
        v16bf b = load_frag(bp, ldb2, k);
        acc = __builtin_amdgcn_wmma_f32_16x16x32_bf16(false, a, false, b,
                                                      (short)0, acc, false, false);
      }
    }
    const int n = n0 + nloc;
    const float bias = bih[n] + bhh[n];
    for (int v = 0; v < 8; ++v) {
      const int m = m0 + mo + v;
      gbuf[(size_t)m * GG + n] = acc[v] + bias + (float)Gx[(size_t)m * GG + n];
    }
  }
  __syncthreads();

  // cell update: i,f,g,o split; 16*512 elems over 512 threads
  for (int i = 0; i < 16; ++i) {
    const int e  = threadIdx.x + i * 512;
    const int br = e >> 9;
    const int j  = e & 511;
    const int b  = m0 + br;
    const float gi = gbuf[(size_t)b * GG + j];
    const float gf = gbuf[(size_t)b * GG + HH  + j];
    const float gg = gbuf[(size_t)b * GG + 2 * HH + j];
    const float go = gbuf[(size_t)b * GG + 3 * HH + j];
    const float c2 = sigf(gf) * cst[(size_t)b * HH + j] + sigf(gi) * tanhf(gg);
    const float h2 = sigf(go) * tanhf(c2);
    cst [(size_t)b * HH + j] = c2;
    hf32[(size_t)b * HH + j] = h2;
    hbf [(size_t)b * ldhb + j] = (bf16_t)h2;
  }
}

// ---------------------------------------------------------------------------
// Attention (one decoder step): scores = enc_proj . h, masked log_softmax
// over s, ctx = sum_s att * enc_hs (att are LOG-probs — matches reference!).
// One block per batch row, 256 threads (one per source position).
// ---------------------------------------------------------------------------
__global__ void __launch_bounds__(256)
attn_step(const float* __restrict__ h,
          const bf16_t* __restrict__ encp,
          const bf16_t* __restrict__ enchs,
          const int* __restrict__ src,
          bf16_t* __restrict__ ctx_out, int ldc)
{
  __shared__ float hs[HH];
  __shared__ float sc[SS];
  __shared__ float red[SS];
  const int b   = blockIdx.x;
  const int tid = threadIdx.x;
  hs[tid]       = h[(size_t)b * HH + tid];
  hs[tid + 256] = h[(size_t)b * HH + 256 + tid];
  __syncthreads();

  const bf16_t* ep = encp + ((size_t)b * SS + tid) * HH;
  float s = 0.0f;
  for (int k = 0; k < HH; ++k) s += (float)ep[k] * hs[k];
  if (src[b * SS + tid] == 0) s = -__builtin_inff();
  sc[tid] = s; red[tid] = s;
  __syncthreads();
  for (int o = 128; o > 0; o >>= 1) {
    if (tid < o) red[tid] = fmaxf(red[tid], red[tid + o]);
    __syncthreads();
  }
  const float mx = red[0];
  __syncthreads();
  red[tid] = expf(sc[tid] - mx);
  __syncthreads();
  for (int o = 128; o > 0; o >>= 1) {
    if (tid < o) red[tid] += red[tid + o];
    __syncthreads();
  }
  const float lse = mx + logf(red[0]);
  __syncthreads();
  sc[tid] -= lse;                 // log-softmax, kept as log-probs
  __syncthreads();

  for (int d0 = 0; d0 < 4; ++d0) {
    const int d = tid + d0 * 256;
    float acc = 0.0f;
    const bf16_t* hp = enchs + (size_t)b * SS * DH + d;
    for (int si = 0; si < SS; ++si) acc += sc[si] * (float)hp[(size_t)si * DH];
    ctx_out[(size_t)b * ldc + d] = (bf16_t)acc;
  }
}

// ------------------------------ small kernels ------------------------------
__global__ void k_convert(const float* __restrict__ in, bf16_t* __restrict__ out, int n) {
  int i = blockIdx.x * 256 + threadIdx.x;
  if (i < n) out[i] = (bf16_t)in[i];
}
__global__ void k_zero(float* __restrict__ p, int n) {
  int i = blockIdx.x * 256 + threadIdx.x;
  if (i < n) p[i] = 0.0f;
}
// out rows indexed (step*BB + b); token = sents[b*sstride + step]
__global__ void k_gather(const int* __restrict__ sents, int sstride,
                         const float* __restrict__ emb, bf16_t* __restrict__ out,
                         int nrows) {
  int idx = blockIdx.x * 256 + threadIdx.x;
  if (idx >= nrows * EE) return;
  int rowi = idx / EE, c = idx % EE;
  int t = rowi >> 6, b = rowi & 63;
  int tok = sents[b * sstride + t];
  out[idx] = (bf16_t)emb[(size_t)tok * EE + c];
}
__global__ void k_pack(const float* __restrict__ a, const float* __restrict__ b,
                       bf16_t* __restrict__ out) {
  int idx = blockIdx.x * 256 + threadIdx.x;
  if (idx >= BB * DH) return;
  int r = idx >> 10, j = idx & 1023;
  float v = (j < HH) ? a[r * HH + j] : b[r * HH + j - HH];
  out[idx] = (bf16_t)v;
}

// ---------------------------------------------------------------------------
extern "C" void kernel_launch(void* const* d_in, const int* in_sizes, int n_in,
                              void* d_out, int out_size, void* d_ws, size_t ws_size,
                              hipStream_t stream)
{
  (void)in_sizes; (void)n_in; (void)out_size; (void)ws_size;
  const int*   src     = (const int*)  d_in[0];
  const int*   tgt     = (const int*)  d_in[1];
  const float* srcEmb  = (const float*)d_in[2];
  const float* tgtEmb  = (const float*)d_in[3];
  const float* eWihF   = (const float*)d_in[4];
  const float* eWhhF   = (const float*)d_in[5];
  const float* ebihF   = (const float*)d_in[6];
  const float* ebhhF   = (const float*)d_in[7];
  const float* eWihB   = (const float*)d_in[8];
  const float* eWhhB   = (const float*)d_in[9];
  const float* ebihB   = (const float*)d_in[10];
  const float* ebhhB   = (const float*)d_in[11];
  const float* WhInit  = (const float*)d_in[12];
  const float* WcInit  = (const float*)d_in[13];
  const float* dWih    = (const float*)d_in[14];
  const float* dWhh    = (const float*)d_in[15];
  const float* dbih    = (const float*)d_in[16];
  const float* dbhh    = (const float*)d_in[17];
  const float* Watt    = (const float*)d_in[18];
  const float* Wcomb   = (const float*)d_in[19];
  const float* Wvocab  = (const float*)d_in[20];

  char* ws = (char*)d_ws;
  size_t off = 0;
  auto alloc = [&](size_t bytes) -> char* {
    char* p = ws + off;
    off += (bytes + 255) & ~(size_t)255;
    return p;
  };

  // bf16 weights
  bf16_t* wEihF = (bf16_t*)alloc((size_t)GG * EE * 2);
  bf16_t* wEhhF = (bf16_t*)alloc((size_t)GG * HH * 2);
  bf16_t* wEihB = (bf16_t*)alloc((size_t)GG * EE * 2);
  bf16_t* wEhhB = (bf16_t*)alloc((size_t)GG * HH * 2);
  bf16_t* wWh   = (bf16_t*)alloc((size_t)HH * DH * 2);
  bf16_t* wWc   = (bf16_t*)alloc((size_t)HH * DH * 2);
  bf16_t* wDih  = (bf16_t*)alloc((size_t)GG * (EE + HH) * 2);
  bf16_t* wDhh  = (bf16_t*)alloc((size_t)GG * HH * 2);
  bf16_t* wAtt  = (bf16_t*)alloc((size_t)HH * DH * 2);
  bf16_t* wCmb  = (bf16_t*)alloc((size_t)HH * 1536 * 2);
  bf16_t* wVoc  = (bf16_t*)alloc((size_t)VT * HH * 2);
  // activations
  bf16_t* Xg    = (bf16_t*)alloc((size_t)SS * BB * EE * 2);
  bf16_t* Xdg   = (bf16_t*)alloc((size_t)TD * BB * EE * 2);
  bf16_t* GxF   = (bf16_t*)alloc((size_t)SS * BB * GG * 2);
  bf16_t* GxB   = (bf16_t*)alloc((size_t)SS * BB * GG * 2);
  bf16_t* GxD   = (bf16_t*)alloc((size_t)TD * BB * GG * 2);
  bf16_t* encHs = (bf16_t*)alloc((size_t)BB * SS * DH * 2);   // [b][s][2H]
  bf16_t* encP  = (bf16_t*)alloc((size_t)BB * SS * HH * 2);   // [b*S+s][H]
  bf16_t* combA = (bf16_t*)alloc((size_t)TD * BB * HH * 2);   // [t*B+b][H]
  bf16_t* Adec  = (bf16_t*)alloc((size_t)BB * 1536 * 2);      // [ctx(1024)|h(512)]
  bf16_t* AinH  = (bf16_t*)alloc((size_t)BB * DH * 2);
  bf16_t* AinC  = (bf16_t*)alloc((size_t)BB * DH * 2);
  float*  hF    = (float*)alloc((size_t)BB * HH * 4);
  float*  cF    = (float*)alloc((size_t)BB * HH * 4);
  float*  hB    = (float*)alloc((size_t)BB * HH * 4);
  float*  cB    = (float*)alloc((size_t)BB * HH * 4);
  float*  decH  = (float*)alloc((size_t)BB * HH * 4);
  float*  decC  = (float*)alloc((size_t)BB * HH * 4);
  float*  gbuf  = (float*)alloc((size_t)BB * GG * 4);

  auto cvt = [&](const float* in, bf16_t* out, int n) {
    k_convert<<<(n + 255) / 256, 256, 0, stream>>>(in, out, n);
  };
  cvt(eWihF, wEihF, GG * EE);   cvt(eWhhF, wEhhF, GG * HH);
  cvt(eWihB, wEihB, GG * EE);   cvt(eWhhB, wEhhB, GG * HH);
  cvt(WhInit, wWh, HH * DH);    cvt(WcInit, wWc, HH * DH);
  cvt(dWih, wDih, GG * (EE + HH)); cvt(dWhh, wDhh, GG * HH);
  cvt(Watt, wAtt, HH * DH);     cvt(Wcomb, wCmb, HH * 1536);
  cvt(Wvocab, wVoc, VT * HH);

  k_gather<<<(SS * BB * EE + 255) / 256, 256, 0, stream>>>(src, SS, srcEmb, Xg, SS * BB);
  k_gather<<<(TD * BB * EE + 255) / 256, 256, 0, stream>>>(tgt, TT, tgtEmb, Xdg, TD * BB);

  // hoisted input-gate GEMMs (time-parallel)
  gemm_bf16<<<dim3(GG / 128, SS * BB / 16), 256, 0, stream>>>(Xg,  EE, wEihF,    EE,  EE, 0, nullptr, 0, GxF, GG);
  gemm_bf16<<<dim3(GG / 128, SS * BB / 16), 256, 0, stream>>>(Xg,  EE, wEihB,    EE,  EE, 0, nullptr, 0, GxB, GG);
  gemm_bf16<<<dim3(GG / 128, TD * BB / 16), 256, 0, stream>>>(Xdg, EE, wDih + HH, EE + HH, EE, 0, nullptr, 0, GxD, GG);

  k_zero<<<(BB * HH + 255) / 256, 256, 0, stream>>>(cF, BB * HH);
  k_zero<<<(BB * HH + 255) / 256, 256, 0, stream>>>(cB, BB * HH);

  // forward encoder scan: h_t written as bf16 into encHs[:, s, 0:H]
  for (int s = 0; s < SS; ++s) {
    const bf16_t* hprev = (s == 0) ? nullptr : encHs + (size_t)(s - 1) * DH;
    lstm_step<<<4, 512, 0, stream>>>(GxF + (size_t)s * BB * GG,
                                     wEhhF, HH, hprev, SS * DH,
                                     nullptr, 0, nullptr, 0,
                                     ebihF, ebhhF, cF, hF,
                                     encHs + (size_t)s * DH, SS * DH, gbuf);
  }
  // backward encoder scan: iterate s = S-1..0, write encHs[:, s, H:2H]
  for (int i = 0; i < SS; ++i) {
    const int s = SS - 1 - i;
    const bf16_t* hprev = (i == 0) ? nullptr : encHs + (size_t)(s + 1) * DH + HH;
    lstm_step<<<4, 512, 0, stream>>>(GxB + (size_t)s * BB * GG,
                                     wEhhB, HH, hprev, SS * DH,
                                     nullptr, 0, nullptr, 0,
                                     ebihB, ebhhB, cB, hB,
                                     encHs + (size_t)s * DH + HH, SS * DH, gbuf);
  }

  // decoder init: dec_h = [hf,hb]@Wh^T (f32 + bf16 into Adec[:,1024:]); dec_c likewise
  k_pack<<<(BB * DH + 255) / 256, 256, 0, stream>>>(hF, hB, AinH);
  k_pack<<<(BB * DH + 255) / 256, 256, 0, stream>>>(cF, cB, AinC);
  gemm_bf16<<<dim3(HH / 128, BB / 16), 256, 0, stream>>>(AinH, DH, wWh, DH, DH, 2, decH, HH, Adec + DH, 1536);
  gemm_bf16<<<dim3(HH / 128, BB / 16), 256, 0, stream>>>(AinC, DH, wWc, DH, DH, 1, decC, HH, nullptr, 0);

  // enc_proj = enc_hs @ W_att^T
  gemm_bf16<<<dim3(HH / 128, BB * SS / 16), 256, 0, stream>>>(encHs, DH, wAtt, DH, DH, 0, nullptr, 0, encP, HH);

  // decoder scan
  for (int t = 0; t < TD; ++t) {
    const bf16_t* oprev = (t == 0) ? nullptr : combA + (size_t)(t - 1) * BB * HH;
    lstm_step<<<4, 512, 0, stream>>>(GxD + (size_t)t * BB * GG,
                                     wDhh, HH, Adec + DH, 1536,
                                     wDih, EE + HH, oprev, HH,
                                     dbih, dbhh, decC, decH,
                                     Adec + DH, 1536, gbuf);
    attn_step<<<BB, 256, 0, stream>>>(decH, encP, encHs, src, Adec, 1536);
    gemm_bf16<<<dim3(HH / 128, BB / 16), 256, 0, stream>>>(Adec, 1536, wCmb, 1536, 1536, 3,
                                                           nullptr, 0,
                                                           combA + (size_t)t * BB * HH, HH);
  }

  // final vocab projection: [TD*B, H] @ W_vocab[V,H]^T -> out[b][t][v]
  gemm_bf16<<<dim3(VT / 128, TD * BB / 16), 256, 0, stream>>>(combA, HH, wVoc, HH, HH, 4,
                                                              (float*)d_out, 0, nullptr, 0);
}
